// PredAttentionValue_85375359910647
// MI455X (gfx1250) — matrix operand order repaired and checked by reading, not compiled, not run on hardware
//
#include <hip/hip_runtime.h>
#include <hip/hip_bf16.h>

typedef _Float16 v16h __attribute__((ext_vector_type(16)));
typedef _Float16 h8   __attribute__((ext_vector_type(8)));
typedef float    v8f  __attribute__((ext_vector_type(8)));

#define BB 32
#define TT 500
#define TP 512          // padded T
#define CC 256
#define HH 8
#define DD 32
#define MROWS (BB * TT) // 16000

static __device__ __forceinline__ v8f wmma_f16(v16h a, v16h b, v8f c) {
    return __builtin_amdgcn_wmma_f32_16x16x32_f16(false, a, false, b, (short)0, c, false, false);
}

// Load a 16-wide A-fragment (row r, K {k0+off..+7, k0+off+16..+23}) from f16 row-major [*, 256].
static __device__ __forceinline__ v16h load_afrag(const _Float16* base, int r, int k0, int hi) {
    v16h a;
    const _Float16* p = base + (size_t)r * CC + k0 + hi * 8;
    *(h8*)&a       = *(const h8*)(p);
    *((h8*)&a + 1) = *(const h8*)(p + 16);
    return a;
}
// Load a B-fragment (col n, K {k0+hi*8..+7, k0+hi*8+16..+23}) from f16 [N][256] (transposed weights).
static __device__ __forceinline__ v16h load_bfrag(const _Float16* Wt, int n, int k0, int hi) {
    v16h b;
    const _Float16* p = Wt + (size_t)n * CC + k0 + hi * 8;
    *(h8*)&b       = *(const h8*)(p);
    *((h8*)&b + 1) = *(const h8*)(p + 16);
    return b;
}

// ---------------------------------------------------------------------------
// prep: transpose weights to f16 [N][K] and convert x to f16 (done once; all
// GEMM operands become straight 16-byte f16 loads with zero inner-loop cvt).
// ---------------------------------------------------------------------------
__global__ __launch_bounds__(256) void prep_kernel(
    const float* __restrict__ Wa, const float* __restrict__ Wp,
    const float* __restrict__ x,
    _Float16* __restrict__ Wta, _Float16* __restrict__ Wtp,
    _Float16* __restrict__ xh)
{
    const int NA = CC * 3 * CC;       // 196608
    const int NP = CC * CC;           // 65536
    int idx = blockIdx.x * 256 + threadIdx.x;
    if (idx < NA) {                                    // W_attn [256][768] -> [768][256]
        int k = idx / (3 * CC), n = idx % (3 * CC);
        Wta[(size_t)n * CC + k] = (_Float16)Wa[idx];
    } else if (idx < NA + NP) {                        // W_proj [256][256] -> [256][256]^T
        int j = idx - NA;
        int k = j / CC, n = j % CC;
        Wtp[(size_t)n * CC + k] = (_Float16)Wp[j];
    } else {                                           // x -> f16, same layout
        int j = idx - NA - NP;                         // j < 4,096,000 (exact grid)
        xh[j] = (_Float16)x[j];
    }
}

// ---------------------------------------------------------------------------
// qkv: qkv = x @ W_attn + b.  One wave = 16(M) x 64(N): one A-fragment feeds
// 4 B-fragments / 4 back-to-back WMMAs per 32-deep k-step (4x A reuse).
// Epilogue scatters bias-added f16 Q,K ([B,H,TP,32]) and V^T ([B,H,32,TP]).
// ---------------------------------------------------------------------------
__global__ __launch_bounds__(128) void qkv_kernel(
    const _Float16* __restrict__ xh, const _Float16* __restrict__ Wt,
    const float* __restrict__ bias,
    _Float16* __restrict__ Qb, _Float16* __restrict__ Kb,
    _Float16* __restrict__ Vt)
{
    const int lane = threadIdx.x & 31;
    const int wid  = threadIdx.x >> 5;
    const int tile = blockIdx.x * 4 + wid;         // 1000 x 12 tiles of 16x64
    const int mt = tile / 12, nt = tile % 12;
    const int ln = lane & 15, hi = lane >> 4;
    const int r  = mt * 16 + ln;
    const int n0 = nt * 64 + ln;

    v8f acc[4];
    for (int u = 0; u < 4; ++u) for (int j = 0; j < 8; ++j) acc[u][j] = 0.f;

    for (int k0 = 0; k0 < CC; k0 += 32) {
        v16h a = load_afrag(xh, r, k0, hi);
        #pragma unroll
        for (int u = 0; u < 4; ++u) {
            v16h bf = load_bfrag(Wt, n0 + u * 16, k0, hi);
            acc[u] = wmma_f16(a, bf, acc[u]);
        }
    }

    // Epilogue: C layout -> lane<16 rows 0..7, lane>=16 rows 8..15, col = ln.
    #pragma unroll
    for (int u = 0; u < 4; ++u) {
        const int nc = nt * 64 + u * 16 + ln;
        const int which = nc >> 8;       // 0=q 1=k 2=v
        const int c  = nc & 255;
        const int hh = c >> 5, dd = c & 31;
        const float bv = bias[nc];
        for (int j = 0; j < 8; ++j) {
            int mr = mt * 16 + hi * 8 + j;       // < 16000 always
            int bb = mr / TT, t = mr % TT;
            _Float16 val = (_Float16)(acc[u][j] + bv);
            size_t bh = (size_t)(bb * HH + hh);
            if (which == 0)      Qb[(bh * TP + t) * DD + dd] = val;
            else if (which == 1) Kb[(bh * TP + t) * DD + dd] = val;
            else                 Vt[(bh * DD + dd) * TP + t] = val;
        }
    }
}

// ---------------------------------------------------------------------------
// attn: one block = (batch b, 16-query tile); wave w = head w (8 waves).
// Flash-style online softmax over 16 key-steps of 32; pred path reduced
// across heads via ds_add_f32 in LDS, then sigmoid -> pred_weight.
// ---------------------------------------------------------------------------
__global__ __launch_bounds__(256) void attn_kernel(
    const _Float16* __restrict__ Q, const _Float16* __restrict__ K,
    const _Float16* __restrict__ Vt, const float* __restrict__ p_w,
    float* __restrict__ pred_out, _Float16* __restrict__ ctx)
{
    __shared__ float predacc[16][32];
    __shared__ __attribute__((aligned(16))) _Float16 pbuf[HH][16][32];

    const int tid  = threadIdx.x;
    const int lane = tid & 31;
    const int h    = tid >> 5;           // wave id == head
    const int b    = blockIdx.x >> 5;
    const int qt   = blockIdx.x & 31;
    const int t0   = qt * 16;
    const int ln   = lane & 15;
    const int hi   = lane >> 4;

    for (int i = tid; i < 16 * 32; i += 256) (&predacc[0][0])[i] = 0.f;
    __syncthreads();

    const float pw = p_w[h];
    const float sscale = 0.17677669529663687f;   // 1/sqrt(32)

    const _Float16* Qh = Q  + (size_t)(b * HH + h) * TP * DD;
    const _Float16* Kh = K  + (size_t)(b * HH + h) * TP * DD;
    const _Float16* Vh = Vt + (size_t)(b * HH + h) * DD * TP;

    // Q A-fragment, reused across all key steps.
    v16h qa;
    {
        const int r = t0 + ln, off = hi * 8;
        *(h8*)&qa       = *(const h8*)(Qh + r * DD + off);
        *((h8*)&qa + 1) = *(const h8*)(Qh + r * DD + off + 16);
    }

    float m[8], lsum[8];
    v8f O0, O1;
    for (int j = 0; j < 8; ++j) { m[j] = -1e30f; lsum[j] = 0.f; O0[j] = 0.f; O1[j] = 0.f; }

    for (int s0 = 0; s0 < TP; s0 += 32) {
        // K^T B-fragments: kd = hi*8, halves at kd and kd+16 (table order).
        v16h kb0, kb1;
        {
            const int kd = hi * 8;
            const _Float16* p0 = Kh + (size_t)(s0 + ln) * DD + kd;
            const _Float16* p1 = Kh + (size_t)(s0 + 16 + ln) * DD + kd;
            *(h8*)&kb0 = *(const h8*)(p0); *((h8*)&kb0 + 1) = *(const h8*)(p0 + 16);
            *(h8*)&kb1 = *(const h8*)(p1); *((h8*)&kb1 + 1) = *(const h8*)(p1 + 16);
        }
        v8f S0, S1;
        for (int j = 0; j < 8; ++j) { S0[j] = 0.f; S1[j] = 0.f; }
        S0 = wmma_f16(qa, kb0, S0);
        S1 = wmma_f16(qa, kb1, S1);
        for (int j = 0; j < 8; ++j) { S0[j] *= sscale; S1[j] *= sscale; }

        // pred path: cross-head weighted sum of raw scaled scores in LDS.
        for (int j = 0; j < 8; ++j) {
            atomicAdd(&predacc[hi * 8 + j][ln],      pw * S0[j]);
            atomicAdd(&predacc[hi * 8 + j][ln + 16], pw * S1[j]);
        }

        // mask padded keys for the softmax path.
        if (s0 + 31 >= TT) {
            if (s0 + ln >= TT)      for (int j = 0; j < 8; ++j) S0[j] = -1e30f;
            if (s0 + 16 + ln >= TT) for (int j = 0; j < 8; ++j) S1[j] = -1e30f;
        }

        // online softmax (rows live in 16-lane halves; xor 1,2,4,8 reduces a row)
        for (int j = 0; j < 8; ++j) {
            float rm = fmaxf(S0[j], S1[j]);
            for (int o = 8; o >= 1; o >>= 1) rm = fmaxf(rm, __shfl_xor(rm, o, 32));
            float mn = fmaxf(m[j], rm);
            float alpha = __expf(m[j] - mn);
            m[j] = mn;
            S0[j] = __expf(S0[j] - mn);
            S1[j] = __expf(S1[j] - mn);
            float rs = S0[j] + S1[j];
            for (int o = 8; o >= 1; o >>= 1) rs += __shfl_xor(rs, o, 32);
            lsum[j] = lsum[j] * alpha + rs;
            O0[j] *= alpha; O1[j] *= alpha;
        }

        // Re-layout P: C-tile -> LDS (f16) -> A-fragment (same-wave, DS in-order).
        for (int j = 0; j < 8; ++j) {
            pbuf[h][hi * 8 + j][ln]      = (_Float16)S0[j];
            pbuf[h][hi * 8 + j][ln + 16] = (_Float16)S1[j];
        }
        asm volatile("s_wait_dscnt 0" ::: "memory");
        v16h pa;
        {
            const int off = hi * 8;
            *(h8*)&pa       = *(const h8*)&pbuf[h][ln][off];
            *((h8*)&pa + 1) = *(const h8*)&pbuf[h][ln][off + 16];
        }

        // V B-fragments from transposed V: keys s0+hi*8+{0..7, 16..23} (table order).
        v16h vb0, vb1;
        {
            const int s = s0 + hi * 8;
            const _Float16* q0 = Vh + (size_t)ln * TP + s;
            const _Float16* q1 = Vh + (size_t)(ln + 16) * TP + s;
            *(h8*)&vb0 = *(const h8*)(q0); *((h8*)&vb0 + 1) = *(const h8*)(q0 + 16);
            *(h8*)&vb1 = *(const h8*)(q1); *((h8*)&vb1 + 1) = *(const h8*)(q1 + 16);
        }
        O0 = wmma_f16(pa, vb0, O0);
        O1 = wmma_f16(pa, vb1, O1);

        // Finalize pred for this 16x32 block: sigmoid + store + reset.
        __syncthreads();
        for (int i = tid; i < 16 * 32; i += 256) {
            int rr = i >> 5, cc = i & 31;
            int t = t0 + rr, s = s0 + cc;
            float v = predacc[rr][cc];
            float sg = 1.f / (1.f + __expf(-v));
            if (isnan(sg)) sg = 0.f;
            if (t < TT && s < TT)
                pred_out[((size_t)b * TT + t) * TT + s] = sg;
            predacc[rr][cc] = 0.f;
        }
        __syncthreads();
    }

    // O /= l ; write ctx f16 [B*T, C] for the projection GEMM.
    for (int j = 0; j < 8; ++j) {
        int t = t0 + hi * 8 + j;
        if (t < TT) {
            float inv = 1.f / lsum[j];
            size_t base = ((size_t)b * TT + t) * CC + h * DD;
            ctx[base + ln]      = (_Float16)(O0[j] * inv);
            ctx[base + ln + 16] = (_Float16)(O1[j] * inv);
        }
    }
}

// ---------------------------------------------------------------------------
// proj: value = ctx @ W_proj + b_proj  (M=16000, K=256, N=256), fp32 out.
// Same 16x64 wave tile as qkv.
// ---------------------------------------------------------------------------
__global__ __launch_bounds__(128) void proj_kernel(
    const _Float16* __restrict__ ctx, const _Float16* __restrict__ Wt,
    const float* __restrict__ bias, float* __restrict__ out)
{
    const int lane = threadIdx.x & 31;
    const int wid  = threadIdx.x >> 5;
    const int tile = blockIdx.x * 4 + wid;          // 1000 x 4 tiles of 16x64
    const int mt = tile / 4, nt = tile % 4;
    const int ln = lane & 15, hi = lane >> 4;
    const int r  = mt * 16 + ln;
    const int n0 = nt * 64 + ln;

    v8f acc[4];
    for (int u = 0; u < 4; ++u) for (int j = 0; j < 8; ++j) acc[u][j] = 0.f;

    for (int k0 = 0; k0 < CC; k0 += 32) {
        v16h a = load_afrag(ctx, r, k0, hi);
        #pragma unroll
        for (int u = 0; u < 4; ++u) {
            v16h bf = load_bfrag(Wt, n0 + u * 16, k0, hi);
            acc[u] = wmma_f16(a, bf, acc[u]);
        }
    }

    #pragma unroll
    for (int u = 0; u < 4; ++u) {
        const int nc = nt * 64 + u * 16 + ln;
        const float bv = bias[nc];
        for (int j = 0; j < 8; ++j) {
            int mr = mt * 16 + hi * 8 + j;
            out[(size_t)mr * CC + nc] = acc[u][j] + bv;
        }
    }
}

// ---------------------------------------------------------------------------
extern "C" void kernel_launch(void* const* d_in, const int* in_sizes, int n_in,
                              void* d_out, int out_size, void* d_ws, size_t ws_size,
                              hipStream_t stream)
{
    (void)in_sizes; (void)n_in; (void)out_size; (void)ws_size;
    const float* x      = (const float*)d_in[0];
    const float* W_attn = (const float*)d_in[1];
    const float* b_attn = (const float*)d_in[2];
    const float* p_w    = (const float*)d_in[3];
    const float* W_proj = (const float*)d_in[4];
    const float* b_proj = (const float*)d_in[5];

    float* pred  = (float*)d_out;                                  // [32,500,500]
    float* value = (float*)d_out + (size_t)BB * TT * TT;           // [32,500,256]

    char* ws = (char*)d_ws;
    _Float16* Wta = (_Float16*)ws; ws += (size_t)3 * CC * CC * 2;  // 768x256 f16
    _Float16* Wtp = (_Float16*)ws; ws += (size_t)CC * CC * 2;      // 256x256 f16
    _Float16* xh  = (_Float16*)ws; ws += (size_t)MROWS * CC * 2;   // x as f16
    _Float16* Qb  = (_Float16*)ws; ws += (size_t)BB * HH * TP * DD * 2;
    _Float16* Kb  = (_Float16*)ws; ws += (size_t)BB * HH * TP * DD * 2;
    _Float16* Vt  = (_Float16*)ws; ws += (size_t)BB * HH * DD * TP * 2;
    _Float16* ctx = (_Float16*)ws; ws += (size_t)BB * TT * CC * 2;

    // Zero Q/K/V (covers the T->512 pad rows so padded keys/values are finite 0).
    hipMemsetAsync(Qb, 0, (size_t)3 * BB * HH * TP * DD * 2, stream);

    // 196608 + 65536 + 4,096,000 = 4,358,144 = 17024 * 256 (exact)
    prep_kernel<<<17024, 256, 0, stream>>>(W_attn, W_proj, x, Wta, Wtp, xh);
    qkv_kernel<<<3000, 128, 0, stream>>>(xh, Wta, b_attn, Qb, Kb, Vt);
    attn_kernel<<<BB * 32, 256, 0, stream>>>(Qb, Kb, Vt, p_w, pred, ctx);
    proj_kernel<<<1000, 128, 0, stream>>>(ctx, Wtp, b_proj, value);
}
